// RCNN_90486370992842
// MI455X (gfx1250) — compile-verified, matrix operand order checked by reading, not run on hardware
//
#include <hip/hip_runtime.h>
#include <hip/hip_fp16.h>

typedef __attribute__((ext_vector_type(16))) _Float16 v16h;
typedef __attribute__((ext_vector_type(8)))  _Float16 v8h;
typedef __attribute__((ext_vector_type(8)))  float    v8f;

#define LDS_STRIDE 40   // 32 f16 data + 8 f16 pad = 80 bytes per LDS row

#if __has_builtin(__builtin_amdgcn_tensor_load_to_lds)
#define USE_TDM 1
#else
#define USE_TDM 0
#endif

// ---- WMMA helpers -------------------------------------------------------

__device__ __forceinline__ v8f wmma16(v16h a, v16h b, v8f c) {
  // D = A(16x32 f16) * B(32x16 f16) + C(16x16 f32)
  return __builtin_amdgcn_wmma_f32_16x16x32_f16(
      /*neg_a=*/false, a, /*neg_b=*/false, b,
      /*c_mod=*/(short)0, c, /*reuse_a=*/false, /*reuse_b=*/false);
}

// Load a 16x32 f16 operand tile (A or B, both row-major along K, row stride
// LDS_STRIDE) into the CDNA5 WMMA VGPR layout:
//   lanes 0-15 : row = lane,  V0-3 = K0..7,  V4-7 = K16..23
//   lanes16-31 : row = l-16,  V0-3 = K8..15, V4-7 = K24..31
__device__ __forceinline__ v16h ldsFrag(const _Float16* base, int lane) {
  int r  = lane & 15;
  int kh = (lane >> 4) * 8;
  const _Float16* p = base + r * LDS_STRIDE + kh;
  v8h lo = *(const v8h*)(p);
  v8h hi = *(const v8h*)(p + 16);
  v16h out;
#pragma unroll
  for (int i = 0; i < 8; ++i) { out[i] = lo[i]; out[i + 8] = hi[i]; }
  return out;
}

#if USE_TDM
// ---- Tensor Data Mover: load a rows x 32-f16 2-D tile into LDS ----------
// Global row stride 1024 f16 (2048 B). LDS rows padded 64B data + 16B pad
// (pad_interval code 3 = 64 B, pad_amount code 3 = 4 DWORDs) -> matches
// LDS_STRIDE = 40 f16 = 80 B.
typedef __attribute__((ext_vector_type(4))) unsigned int u32x4;
typedef __attribute__((ext_vector_type(8))) int i32x8;
typedef __attribute__((ext_vector_type(4))) int i32x4;

__device__ __forceinline__ void tdm_load_tile(unsigned lds_off,
                                              const _Float16* g,
                                              unsigned rows) {
  unsigned long long ga = (unsigned long long)(size_t)g;
  u32x4 g0 = {};
  g0[0] = 1u;                                        // count=1, user mode
  g0[1] = lds_off;                                   // D#.lds_addr
  g0[2] = (unsigned)ga;                              // global_addr[31:0]
  g0[3] = (unsigned)((ga >> 32) & 0x01ffffffu) | (2u << 30);  // ga[56:32], type=2
  i32x8 g1 = {};
  g1[0] = (int)((1u << 16)    // data_size = 2 bytes
              | (1u << 20)    // pad_enable
              | (3u << 22)    // pad_interval: 64 B between pads
              | (3u << 25));  // pad_amount: 4 DWORDs (16 B)
  g1[1] = (int)(1024u << 16); // tensor_dim0 = 1024 (lo16 in [63:48])
  g1[2] = 0;                  // tensor_dim0 hi = 0, tensor_dim1 lo16 = 0
  g1[3] = (int)(16u | (32u << 16)); // tensor_dim1 = 1<<20 (hi16), tile_dim0 = 32
  g1[4] = (int)rows;          // tile_dim1 = rows, tile_dim2 = 0
  g1[5] = 1024;               // tensor_dim0_stride = 1024 elements
  g1[6] = 0;
  g1[7] = 0;
  i32x4 z4 = {};
  i32x8 z8 = {};
  // 6-arg toolchain variant (clang-23): groups 0..3, aux group, cache policy
  __builtin_amdgcn_tensor_load_to_lds(g0, g1, z4, z4, z8, 0);
}
#endif

// ---- Conversion / layout kernels ---------------------------------------

__global__ void k_f32_to_f16(const float* __restrict__ in, _Float16* __restrict__ out, int n) {
  int stride = gridDim.x * blockDim.x;
  for (int i = blockIdx.x * blockDim.x + threadIdx.x; i < n; i += stride)
    out[i] = (_Float16)in[i];
}

// in (R, C) f32 -> out (R, Cp) f16, zero pad c >= C
__global__ void k_padcols(const float* __restrict__ in, _Float16* __restrict__ out,
                          int R, int C, int Cp) {
  int stride = gridDim.x * blockDim.x;
  int n = R * Cp;
  for (int i = blockIdx.x * blockDim.x + threadIdx.x; i < n; i += stride) {
    int r = i / Cp, c = i - r * Cp;
    out[i] = (c < C) ? (_Float16)in[r * C + c] : (_Float16)0.f;
  }
}

// in (R, C) f32 -> out (Cp, R) f16 transposed, zero pad rows c >= C
__global__ void k_transpose(const float* __restrict__ in, _Float16* __restrict__ out,
                            int R, int C, int Cp) {
  int stride = gridDim.x * blockDim.x;
  int n = Cp * R;
  for (int i = blockIdx.x * blockDim.x + threadIdx.x; i < n; i += stride) {
    int c = i / R, r = i - c * R;
    out[i] = (c < C) ? (_Float16)in[r * C + c] : (_Float16)0.f;
  }
}

// bc[n] = b2[n] + sum_j b1[j] * w2[j, n]   (fp32, tiny)
__global__ void k_bias_combine(const float* __restrict__ b1, const float* __restrict__ w2,
                               const float* __restrict__ b2, float* __restrict__ bc,
                               int C, int Cp) {
  int n = threadIdx.x;
  if (n >= Cp) return;
  float s = 0.f;
  if (n < C) {
    s = b2[n];
    for (int j = 0; j < 1024; ++j) s += b1[j] * w2[j * C + n];
  }
  bc[n] = s;
}

// ---- Conv as WMMA GEMM: feats[b,f,l] = relu(im2col(sent) @ Wconv^T + b) --
// M = 4096 (b*128+l), N = 128 (f), K = 928 (900 padded). Tile 128x128.
__global__ __launch_bounds__(256) void k_conv_gemm(
    const _Float16* __restrict__ sent,   // (32, 128*300) f16
    const _Float16* __restrict__ wconv,  // (128, 928) f16, (out,in) layout
    const float*    __restrict__ bias,   // (128,) f32
    _Float16* __restrict__ feats)        // (32, 128, 128) f16
{
  __shared__ _Float16 As[128 * LDS_STRIDE];
  __shared__ _Float16 Bs[128 * LDS_STRIDE];
  const int tid = threadIdx.x, wave = tid >> 5, lane = tid & 31;
  const int m0 = blockIdx.x * 128;

  v8f zacc = {};
  v8f c[8];
#pragma unroll
  for (int i = 0; i < 8; ++i) c[i] = zacc;

  for (int kb = 0; kb < 928; kb += 32) {
    // A tile: im2col with edge masking (row m=(b,l): k maps to (l-1)*300+k)
    for (int e = tid; e < 128 * 32; e += 256) {
      int r = e >> 5, cc = e & 31;
      int m = m0 + r;
      int b = m >> 7, l = m & 127;
      int k = kb + cc;
      int off = (l - 1) * 300 + k;
      _Float16 v = (_Float16)0.f;
      if (k < 900 && off >= 0 && off < 38400) v = sent[b * 38400 + off];
      As[r * LDS_STRIDE + cc] = v;
    }
    // B tile: 128 rows x 32 cols, vectorized 16B chunks
#pragma unroll
    for (int it = 0; it < 2; ++it) {
      int ch = tid + it * 256;
      int r = ch >> 2, c8 = (ch & 3) * 8;
      *(v8h*)(&Bs[r * LDS_STRIDE + c8]) = *(const v8h*)(wconv + r * 928 + kb + c8);
    }
    __syncthreads();
    v16h a = ldsFrag(&As[(wave * 16) * LDS_STRIDE], lane);
#pragma unroll
    for (int nt = 0; nt < 8; ++nt) {
      v16h b = ldsFrag(&Bs[(nt * 16) * LDS_STRIDE], lane);
      c[nt] = wmma16(a, b, c[nt]);
    }
    __syncthreads();
  }
  // epilogue: +bias, ReLU, transposed store as (b, f, l)
  int nloc = lane & 15, mh = (lane >> 4) * 8;
#pragma unroll
  for (int nt = 0; nt < 8; ++nt) {
    int f = nt * 16 + nloc;
    float bv = bias[f];
#pragma unroll
    for (int r = 0; r < 8; ++r) {
      int m = m0 + wave * 16 + mh + r;
      float v = c[nt][r] + bv;
      v = v > 0.f ? v : 0.f;
      int b = m >> 7, l = m & 127;
      feats[(b * 128 + f) * 128 + l] = (_Float16)v;
    }
  }
}

// ---- ROI max pool: res[n, f*8+p] ---------------------------------------
__global__ __launch_bounds__(256) void k_roipool(
    const _Float16* __restrict__ feats,  // (32,128,128)
    const int* __restrict__ rois,        // (4096,2)
    const int* __restrict__ ridx,        // (4096,)
    _Float16* __restrict__ res)          // (4096,1024)
{
  int n  = blockIdx.x;
  int x1 = rois[2 * n];
  int x2 = rois[2 * n + 1] + 1;
  int lr = x2 - x1;
  const _Float16* fb = feats + ridx[n] * (128 * 128);
  for (int t = threadIdx.x; t < 1024; t += 256) {
    int f = t >> 3, p = t & 7;
    int start = x1 + (p * lr) / 8;
    int end   = x1 + ((p + 1) * lr + 7) / 8;
    const _Float16* row = fb + f * 128;
    float mx = -__builtin_huge_valf();
    for (int w = 0; w < 18; ++w) {           // MAXW = 128/8 + 2
      int idx = start + w;
      if (idx < end) {
        int ic = idx < 127 ? idx : 127;
        float v = (float)row[ic];
        mx = v > mx ? v : mx;
      }
    }
    res[n * 1024 + f * 8 + p] = (_Float16)mx;
  }
}

// ---- Combine: WcT = (W1 @ W2)^T  via WMMA ------------------------------
// A = W1 (1024 k-rows x 1024 j-cols, row-major), B = W2T (NT*16 n-rows x 1024 j)
__global__ __launch_bounds__(256) void k_combine(
    const _Float16* __restrict__ w1, const _Float16* __restrict__ w2t,
    _Float16* __restrict__ wct, int NT)
{
  __shared__ _Float16 As[128 * LDS_STRIDE];
  __shared__ _Float16 Bs[32 * LDS_STRIDE];
  const int tid = threadIdx.x, wave = tid >> 5, lane = tid & 31;
  const int m0 = blockIdx.x * 128;
  v8f zacc = {};
  v8f c[2]; c[0] = zacc; c[1] = zacc;
  const int nrow = NT * 16;
  for (int kb = 0; kb < 1024; kb += 32) {
#pragma unroll
    for (int it = 0; it < 2; ++it) {
      int ch = tid + it * 256;
      int r = ch >> 2, c8 = (ch & 3) * 8;
      *(v8h*)(&As[r * LDS_STRIDE + c8]) = *(const v8h*)(w1 + (m0 + r) * 1024 + kb + c8);
    }
    if (tid < nrow * 4) {
      int r = tid >> 2, c8 = (tid & 3) * 8;
      *(v8h*)(&Bs[r * LDS_STRIDE + c8]) = *(const v8h*)(w2t + r * 1024 + kb + c8);
    }
    __syncthreads();
    v16h a = ldsFrag(&As[(wave * 16) * LDS_STRIDE], lane);
    for (int nt = 0; nt < NT; ++nt) {
      v16h b = ldsFrag(&Bs[(nt * 16) * LDS_STRIDE], lane);
      c[nt] = wmma16(a, b, c[nt]);
    }
    __syncthreads();
  }
  int nloc = lane & 15, mh = (lane >> 4) * 8;
  for (int nt = 0; nt < NT; ++nt) {
    int ncol = nt * 16 + nloc;
#pragma unroll
    for (int r = 0; r < 8; ++r) {
      int m = m0 + wave * 16 + mh + r;
      wct[ncol * 1024 + m] = (_Float16)c[nt][r];   // store transposed -> (n, k)
    }
  }
}

// ---- Head GEMM: out = res @ WcT^T + bc, fp32 out, masked narrow N -------
// TDM path: wave 0 issues double-buffered tensor_load_to_lds for the A
// (64x32) and B (nrow x 32) tiles; LDS padding done by the TDM pad engine.
__global__ __launch_bounds__(128) void k_head_gemm(
    const _Float16* __restrict__ res,  // (4096, 1024)
    const _Float16* __restrict__ wct,  // (NT*16, 1024)
    const float* __restrict__ bc,      // (NT*16,)
    float* __restrict__ out,           // (4096, ncol)
    int NT, int ncol)
{
  const int tid = threadIdx.x, wave = tid >> 5, lane = tid & 31;
  const int m0 = blockIdx.x * 64;
  v8f zacc = {};
  v8f c[2]; c[0] = zacc; c[1] = zacc;
  const int nrow = NT * 16;

#if USE_TDM
  __shared__ _Float16 As[2][64 * LDS_STRIDE];
  __shared__ _Float16 Bs[2][32 * LDS_STRIDE];
  unsigned aoff[2] = { (unsigned)(size_t)&As[0][0], (unsigned)(size_t)&As[1][0] };
  unsigned boff[2] = { (unsigned)(size_t)&Bs[0][0], (unsigned)(size_t)&Bs[1][0] };

  if (wave == 0) {
    tdm_load_tile(aoff[0], res + (size_t)m0 * 1024, 64u);
    tdm_load_tile(boff[0], wct, (unsigned)nrow);
  }
  for (int it = 0; it < 32; ++it) {
    int buf = it & 1;
    if (wave == 0) {
      if (it + 1 < 32) {
        int kb = (it + 1) * 32;
        tdm_load_tile(aoff[buf ^ 1], res + (size_t)m0 * 1024 + kb, 64u);
        tdm_load_tile(boff[buf ^ 1], wct + kb, (unsigned)nrow);
        __builtin_amdgcn_s_wait_tensorcnt(2);  // current buffer's pair done
      } else {
        __builtin_amdgcn_s_wait_tensorcnt(0);
      }
    }
    __syncthreads();                            // publish buf to all waves
    v16h a = ldsFrag(&As[buf][(wave * 16) * LDS_STRIDE], lane);
    for (int nt = 0; nt < NT; ++nt) {
      v16h b = ldsFrag(&Bs[buf][(nt * 16) * LDS_STRIDE], lane);
      c[nt] = wmma16(a, b, c[nt]);
    }
    __syncthreads();   // buf fully consumed before it is re-targeted (it+2)
  }
#else
  __shared__ _Float16 As[64 * LDS_STRIDE];
  __shared__ _Float16 Bs[32 * LDS_STRIDE];
  for (int kb = 0; kb < 1024; kb += 32) {
#pragma unroll
    for (int it = 0; it < 2; ++it) {
      int ch = tid + it * 128;
      int r = ch >> 2, c8 = (ch & 3) * 8;
      *(v8h*)(&As[r * LDS_STRIDE + c8]) = *(const v8h*)(res + (size_t)(m0 + r) * 1024 + kb + c8);
    }
    if (tid < nrow * 4) {
      int r = tid >> 2, c8 = (tid & 3) * 8;
      *(v8h*)(&Bs[r * LDS_STRIDE + c8]) = *(const v8h*)(wct + r * 1024 + kb + c8);
    }
    __syncthreads();
    v16h a = ldsFrag(&As[(wave * 16) * LDS_STRIDE], lane);
    for (int nt = 0; nt < NT; ++nt) {
      v16h b = ldsFrag(&Bs[(nt * 16) * LDS_STRIDE], lane);
      c[nt] = wmma16(a, b, c[nt]);
    }
    __syncthreads();
  }
#endif

  int nloc = lane & 15, mh = (lane >> 4) * 8;
  for (int nt = 0; nt < NT; ++nt) {
    int ncidx = nt * 16 + nloc;
    if (ncidx < ncol) {
      float bv = bc[ncidx];
#pragma unroll
      for (int r = 0; r < 8; ++r) {
        int m = m0 + wave * 16 + mh + r;
        out[m * ncol + ncidx] = c[nt][r] + bv;
      }
    }
  }
}

// ---- Host launch --------------------------------------------------------

extern "C" void kernel_launch(void* const* d_in, const int* in_sizes, int n_in,
                              void* d_out, int out_size, void* d_ws, size_t ws_size,
                              hipStream_t stream) {
  const float* sentence = (const float*)d_in[0];
  const int*   rois     = (const int*)  d_in[1];
  const int*   ridx     = (const int*)  d_in[2];
  const float* conv_w   = (const float*)d_in[3];
  const float* conv_b   = (const float*)d_in[4];
  const float* cls_w1   = (const float*)d_in[5];
  const float* cls_b1   = (const float*)d_in[6];
  const float* cls_w2   = (const float*)d_in[7];
  const float* cls_b2   = (const float*)d_in[8];
  const float* bbox_w1  = (const float*)d_in[9];
  const float* bbox_b1  = (const float*)d_in[10];
  const float* bbox_w2  = (const float*)d_in[11];
  const float* bbox_b2  = (const float*)d_in[12];
  float* out = (float*)d_out;

  // workspace carve-up (all offsets 256B aligned)
  uint8_t* w = (uint8_t*)d_ws;
  size_t off = 0;
  auto alloc = [&](size_t bytes) { uint8_t* p = w + off; off += (bytes + 255) & ~(size_t)255; return p; };
  _Float16* sent16   = (_Float16*)alloc(1228800 * 2);     // (32, 128*300)
  _Float16* convw16  = (_Float16*)alloc(128 * 928 * 2);   // (128, 928)
  _Float16* w1c16    = (_Float16*)alloc(1048576 * 2);     // cls_w1 f16
  _Float16* w1b16    = (_Float16*)alloc(1048576 * 2);     // bbox_w1 f16
  _Float16* w2ct16   = (_Float16*)alloc(16 * 1024 * 2);   // cls_w2^T padded
  _Float16* w2bt16   = (_Float16*)alloc(32 * 1024 * 2);   // bbox_w2^T padded
  _Float16* wct_cls  = (_Float16*)alloc(16 * 1024 * 2);   // (W1@W2)^T cls
  _Float16* wct_bbox = (_Float16*)alloc(32 * 1024 * 2);   // (W1@W2)^T bbox
  float*    bc_cls   = (float*)   alloc(16 * 4);
  float*    bc_bbox  = (float*)   alloc(32 * 4);
  _Float16* feats16  = (_Float16*)alloc(32 * 128 * 128 * 2); // (B,F,L)
  _Float16* res16    = (_Float16*)alloc(4096 * 1024 * 2);    // pooled
  (void)ws_size; (void)in_sizes; (void)n_in; (void)out_size;

  // 1) precision / layout prep
  k_f32_to_f16<<<1024, 256, 0, stream>>>(sentence, sent16, 1228800);
  k_padcols  <<<512, 256, 0, stream>>>(conv_w, convw16, 128, 900, 928);
  k_f32_to_f16<<<1024, 256, 0, stream>>>(cls_w1,  w1c16, 1048576);
  k_f32_to_f16<<<1024, 256, 0, stream>>>(bbox_w1, w1b16, 1048576);
  k_transpose<<<64, 256, 0, stream>>>(cls_w2,  w2ct16, 1024, 11, 16);
  k_transpose<<<128, 256, 0, stream>>>(bbox_w2, w2bt16, 1024, 22, 32);
  k_bias_combine<<<1, 16, 0, stream>>>(cls_b1,  cls_w2,  cls_b2,  bc_cls,  11, 16);
  k_bias_combine<<<1, 32, 0, stream>>>(bbox_b1, bbox_w2, bbox_b2, bc_bbox, 22, 32);

  // 2) collapse the two (activation-free) linear layers: WcT = (W1@W2)^T
  k_combine<<<8, 256, 0, stream>>>(w1c16, w2ct16, wct_cls, 1);
  k_combine<<<8, 256, 0, stream>>>(w1b16, w2bt16, wct_bbox, 2);

  // 3) conv as WMMA GEMM (+bias+ReLU), feats stored (B,F,L)
  k_conv_gemm<<<32, 256, 0, stream>>>(sent16, convw16, conv_b, feats16);

  // 4) ROI max pool -> res (4096,1024) f16
  k_roipool<<<4096, 256, 0, stream>>>(feats16, rois, ridx, res16);

  // 5) heads: cls (4096x11) then bbox (4096x22) straight into d_out
  k_head_gemm<<<64, 128, 0, stream>>>(res16, wct_cls,  bc_cls,  out,             1, 11);
  k_head_gemm<<<64, 128, 0, stream>>>(res16, wct_bbox, bc_bbox, out + 4096 * 11, 2, 22);
}
